// MSA_Attn_80711025426890
// MI455X (gfx1250) — matrix operand order repaired and checked by reading, not compiled, not run on hardware
//
#include <hip/hip_runtime.h>
#include <hip/hip_bf16.h>
#include <math.h>

// ---------------- problem constants ----------------
#define CD   256
#define GD   8
#define PPG  32
#define DD   128
#define LD   384
#define TD   (DD * LD)          // 49152 tokens

typedef __attribute__((ext_vector_type(16))) __bf16 v16bf;
typedef __attribute__((ext_vector_type(8)))  float  v8f;

__device__ __forceinline__ unsigned short f2bf(float f) {
    unsigned u = __float_as_uint(f);
    u += 0x7FFFu + ((u >> 16) & 1u);          // round-to-nearest-even
    return (unsigned short)(u >> 16);
}

union FragU { uint4 q[2]; v16bf v; };

// ---------------- generic batched bf16 WMMA GEMM (32x32 wave tiles) ----------------
// C[b][M][N] = act( scale * A[b][M][K] * B[b][N][K]^T + bias ) + add
// A, B are bf16 (B is stored transposed: N rows of K), C is fp32 or bf16.
// Each wave computes a 32x32 tile as 2x2 WMMA 16x16 tiles with register reuse:
// 2 A-frags + 2 B-frags -> 4 v_wmma per K-step (2 b128 loads per WMMA).
// Two-level batching: blockIdx.y = b1*b2 + bi  (offsets b1*s?1 + bi*s?2).
struct GemmP {
    const unsigned short* A;
    const unsigned short* B;
    void*        C;
    const float* bias;   // length N, or null
    const float* add;    // fp32 residual, same layout as C, or null
    long sA1, sA2, sB1, sB2, sC1, sC2, sAdd1, sAdd2;
    int  M, N, K, lda, ldb, ldc, act, b2;   // act: 0 none, 1 sigmoid, 2 relu
    float scale;
};

template <int OUT_BF16>
__global__ void k_gemm(GemmP p) {
    const int wave = threadIdx.x >> 5;
    const int lane = threadIdx.x & 31;
    const int half = lane >> 4;         // K-half selector for A/B fragments
    const int l16  = lane & 15;
    const int tnCnt  = p.N >> 5;                    // 32-wide tiles
    const int ntiles = (p.M >> 5) * tnCnt;
    const int tile   = blockIdx.x * 8 + wave;       // wave-uniform guard
    if (tile >= ntiles) return;
    const int tm = tile / tnCnt;
    const int tn = tile - tm * tnCnt;
    const int b1 = blockIdx.y / p.b2;
    const int bi = blockIdx.y - b1 * p.b2;

    const unsigned short* Abase = p.A + b1 * p.sA1 + bi * p.sA2;
    const unsigned short* Bbase = p.B + b1 * p.sB1 + bi * p.sB2;
    const unsigned short* Arow0 = Abase + (long)(tm * 32 + l16) * p.lda;
    const unsigned short* Arow1 = Arow0 + 16L * p.lda;
    const unsigned short* Brow0 = Bbase + (long)(tn * 32 + l16) * p.ldb;
    const unsigned short* Brow1 = Brow0 + 16L * p.ldb;

    v8f acc00, acc01, acc10, acc11;
#pragma unroll
    for (int i = 0; i < 8; ++i) { acc00[i] = 0.f; acc01[i] = 0.f; acc10[i] = 0.f; acc11[i] = 0.f; }

    // A frag per lane: row; elems 0..7 = K[k0+half*8 .. +7],
    //                  elems 8..15 = K[k0+half*8+16 .. +23]  (ISA 16-bit A 16x32)
    // B frag per lane: row of B^T (col of B); same K order as A frag
    for (int k0 = 0; k0 < p.K; k0 += 32) {
        FragU a0, a1, b0, b1;
        a0.q[0] = *(const uint4*)(const void*)(Arow0 + k0 + half * 8);
        a0.q[1] = *(const uint4*)(const void*)(Arow0 + k0 + half * 8 + 16);
        a1.q[0] = *(const uint4*)(const void*)(Arow1 + k0 + half * 8);
        a1.q[1] = *(const uint4*)(const void*)(Arow1 + k0 + half * 8 + 16);
        b0.q[0] = *(const uint4*)(const void*)(Brow0 + k0 + half * 8);
        b0.q[1] = *(const uint4*)(const void*)(Brow0 + k0 + half * 8 + 16);
        b1.q[0] = *(const uint4*)(const void*)(Brow1 + k0 + half * 8);
        b1.q[1] = *(const uint4*)(const void*)(Brow1 + k0 + half * 8 + 16);
        acc00 = __builtin_amdgcn_wmma_f32_16x16x32_bf16(
            false, a0.v, false, b0.v, (short)0, acc00, false, false);
        acc01 = __builtin_amdgcn_wmma_f32_16x16x32_bf16(
            false, a0.v, false, b1.v, (short)0, acc01, false, false);
        acc10 = __builtin_amdgcn_wmma_f32_16x16x32_bf16(
            false, a1.v, false, b0.v, (short)0, acc10, false, false);
        acc11 = __builtin_amdgcn_wmma_f32_16x16x32_bf16(
            false, a1.v, false, b1.v, (short)0, acc11, false, false);
    }

    const long cbase = b1 * p.sC1 + bi * p.sC2;
    const long abase = b1 * p.sAdd1 + bi * p.sAdd2;
#pragma unroll
    for (int ai = 0; ai < 2; ++ai) {
#pragma unroll
        for (int bj = 0; bj < 2; ++bj) {
            const v8f acc = (ai == 0) ? (bj == 0 ? acc00 : acc01)
                                      : (bj == 0 ? acc10 : acc11);
            const int n = tn * 32 + bj * 16 + l16;
            const float bv = p.bias ? p.bias[n] : 0.f;
#pragma unroll
            for (int r = 0; r < 8; ++r) {
                const int m = tm * 32 + ai * 16 + half * 8 + r;  // C/D: VGPR r, half
                float v = acc[r] * p.scale + bv;
                if (p.act == 1)      v = 1.f / (1.f + __expf(-v));
                else if (p.act == 2) v = fmaxf(v, 0.f);
                if (p.add) v += p.add[abase + (long)m * p.ldc + n];
                if (OUT_BF16)
                    ((unsigned short*)p.C)[cbase + (long)m * p.ldc + n] = f2bf(v);
                else
                    ((float*)p.C)[cbase + (long)m * p.ldc + n] = v;
            }
        }
    }
}

// ---------------- weight prep: fp32 KxN -> bf16 NxK (transposed) ----------------
__global__ void k_wprep(const float* __restrict__ W, unsigned short* __restrict__ Wt,
                        int K, int N) {
    const long total = (long)K * N;
    for (long idx = blockIdx.x * 256L + threadIdx.x; idx < total;
         idx += (long)gridDim.x * 256L) {
        const int n = (int)(idx % N), k = (int)(idx / N);
        Wt[(long)n * K + k] = f2bf(W[idx]);
    }
}

// ---------------- layernorm over last dim (256), bf16 out ----------------
__global__ void k_ln_bf16(const float* __restrict__ x, const float* __restrict__ g,
                          const float* __restrict__ b, unsigned short* __restrict__ out) {
    __shared__ float red[256];
    const long row = blockIdx.x;
    const int  tid = threadIdx.x;
    float v = x[row * 256 + tid];
    red[tid] = v; __syncthreads();
    for (int s = 128; s > 0; s >>= 1) { if (tid < s) red[tid] += red[tid + s]; __syncthreads(); }
    const float mean = red[0] * (1.f / 256.f); __syncthreads();
    const float d = v - mean;
    red[tid] = d * d; __syncthreads();
    for (int s = 128; s > 0; s >>= 1) { if (tid < s) red[tid] += red[tid + s]; __syncthreads(); }
    const float rstd = rsqrtf(red[0] * (1.f / 256.f) + 1e-5f);
    out[row * 256 + tid] = f2bf(d * rstd * g[tid] + b[tid]);
}

// ---------------- layernorm(bias) @ Bw(256x8) -> Pb[g][l][m] ----------------
__global__ void k_ln_biasproj(const float* __restrict__ bias, const float* __restrict__ g,
                              const float* __restrict__ b, const float* __restrict__ Bw,
                              float* __restrict__ out) {
    __shared__ float red[256];
    __shared__ float vals[256];
    const long row = blockIdx.x;            // l*LD + m
    const int  tid = threadIdx.x;
    float v = bias[row * 256 + tid];
    red[tid] = v; __syncthreads();
    for (int s = 128; s > 0; s >>= 1) { if (tid < s) red[tid] += red[tid + s]; __syncthreads(); }
    const float mean = red[0] * (1.f / 256.f); __syncthreads();
    const float d = v - mean;
    red[tid] = d * d; __syncthreads();
    for (int s = 128; s > 0; s >>= 1) { if (tid < s) red[tid] += red[tid + s]; __syncthreads(); }
    const float rstd = rsqrtf(red[0] * (1.f / 256.f) + 1e-5f);
    vals[tid] = d * rstd * g[tid] + b[tid];
    __syncthreads();
    const int gg = tid >> 5, j = tid & 31;  // 8 wave-groups, one output each
    float part = 0.f;
    for (int ch = j; ch < 256; ch += 32) part += vals[ch] * Bw[ch * 8 + gg];
    red[tid] = part; __syncthreads();
    for (int s = 16; s > 0; s >>= 1) { if (j < s) red[tid] += red[tid + s]; __syncthreads(); }
    if (j == 0) out[(long)gg * (LD * LD) + row] = red[tid];
}

// ---------------- row-softmax (fp32 in, bf16 out) ----------------
__global__ void k_softmax_bf16(const float* __restrict__ in,
                               unsigned short* __restrict__ out, int R) {
    __shared__ float red[256];
    const long row = blockIdx.x;
    const int  tid = threadIdx.x;
    const float* p = in + row * (long)R;
    float mx = -3.4e38f;
    for (int i = tid; i < R; i += 256) mx = fmaxf(mx, p[i]);
    red[tid] = mx; __syncthreads();
    for (int s = 128; s > 0; s >>= 1) { if (tid < s) red[tid] = fmaxf(red[tid], red[tid + s]); __syncthreads(); }
    mx = red[0]; __syncthreads();
    float sum = 0.f;
    for (int i = tid; i < R; i += 256) sum += __expf(p[i] - mx);
    red[tid] = sum; __syncthreads();
    for (int s = 128; s > 0; s >>= 1) { if (tid < s) red[tid] += red[tid + s]; __syncthreads(); }
    const float inv = 1.f / red[0];
    for (int i = tid; i < R; i += 256)
        out[row * (long)R + i] = f2bf(__expf(p[i] - mx) * inv);
}

// ---------------- rearrange fp32 [t=(d,l)][ch=(c,g)] -> bf16 attention layouts -----
// mode 0: [g][l][d][c]   (Q/K for row & col logits)
// mode 1: [g][d*32+c][l] (row V^T for attn@V)
// mode 2: [g][l][c][d]   (col V^T for attn@V)
__global__ void k_rearrange(const float* __restrict__ in, unsigned short* __restrict__ out,
                            float scale, int mode) {
    const long total = (long)TD * 256;
    for (long idx = blockIdx.x * 256L + threadIdx.x; idx < total;
         idx += (long)gridDim.x * 256L) {
        const int  ch = (int)(idx & 255);
        const long t  = idx >> 8;
        const int  l  = (int)(t % LD), d = (int)(t / LD);
        const int  c  = ch >> 3, g = ch & 7;
        long dst;
        if (mode == 0)      dst = (((long)g * LD + l) * DD + d) * PPG + c;
        else if (mode == 1) dst = ((long)g * 4096 + d * 32 + c) * LD + l;
        else                dst = (((long)g * LD + l) * PPG + c) * DD + d;
        out[dst] = f2bf(in[idx] * scale);
    }
}

// ---------------- gate * attnOut, scatter back to [t][ch], bf16 ----------------
__global__ void k_gate_mul(const float* __restrict__ gate, const float* __restrict__ O,
                           unsigned short* __restrict__ out) {
    const long total = (long)TD * 256;
    for (long idx = blockIdx.x * 256L + threadIdx.x; idx < total;
         idx += (long)gridDim.x * 256L) {
        const int  ch = (int)(idx & 255);
        const long t  = idx >> 8;
        const int  l  = (int)(t % LD), d = (int)(t / LD);
        const int  c  = ch >> 3, g = ch & 7;
        const long src = (((long)g * LD + l) * DD + d) * PPG + c;  // [g][l][d][c]
        out[idx] = f2bf(gate[idx] * O[src]);
    }
}

// ---------------- workspace layout (bytes) ----------------
static const size_t OFF_WT = 0;                       // bf16 transposed weights
static const size_t OFF_A  = 2359296;                 // fp32 T*256 scratch (proj)
static const size_t OFF_BQ = 52690944;                // bf16 8*384*4096
static const size_t OFF_BK = 77856768;
static const size_t OFF_BV = 103022592;
static const size_t OFF_GT = 128188416;               // fp32 T*256 gate
static const size_t OFF_PB = 178520064;               // fp32 8*384*384 pair bias
static const size_t OFF_LG = 183238656;               // fp32 logits (max 8*48*128*128)
static const size_t OFF_AB = 208404480;               // bf16 softmaxed attn
static const size_t OFF_OV = 220987392;               // fp32 8*384*4096 attn@V
static const size_t OFF_OG = 271319040;               // bf16 T*256 gated
static const size_t OFF_X1 = 296484864;               // fp32 T*256
static const size_t OFF_X2 = 346816512;               // fp32 T*256
static const size_t OFF_XN = 397148160;               // bf16 T*256
static const size_t WS_NEED = 422313984;
// FFN hidden (bf16 T*1024 = 100663296 B) aliases OFF_A..OFF_BV (dead in FFN stage).

extern "C" void kernel_launch(void* const* d_in, const int* in_sizes, int n_in,
                              void* d_out, int out_size, void* d_ws, size_t ws_size,
                              hipStream_t stream) {
    if (ws_size < WS_NEED) return;

    // ---- input pointers (setup_inputs() dict order, recursive) ----
    const float* x      = (const float*)d_in[0];
    const float* biasT  = (const float*)d_in[1];
    const float* rKw    = (const float*)d_in[2];
    const float* rQw    = (const float*)d_in[3];
    const float* rVw    = (const float*)d_in[4];
    const float* rBw    = (const float*)d_in[5];
    const float* rOw    = (const float*)d_in[6];
    const float* rOb    = (const float*)d_in[7];
    const float* rGw    = (const float*)d_in[8];
    const float* rGb    = (const float*)d_in[9];
    const float* rLg    = (const float*)d_in[10];
    const float* rLb    = (const float*)d_in[11];
    const float* rLbg   = (const float*)d_in[12];
    const float* rLbb   = (const float*)d_in[13];
    const float* cKw    = (const float*)d_in[14];
    const float* cQw    = (const float*)d_in[15];
    const float* cVw    = (const float*)d_in[16];
    const float* cOw    = (const float*)d_in[17];
    const float* cOb    = (const float*)d_in[18];
    const float* cGw    = (const float*)d_in[19];
    const float* cGb    = (const float*)d_in[20];
    const float* cLg    = (const float*)d_in[21];
    const float* cLb    = (const float*)d_in[22];
    const float* fW1    = (const float*)d_in[23];
    const float* fB1    = (const float*)d_in[24];
    const float* fW2    = (const float*)d_in[25];
    const float* fB2    = (const float*)d_in[26];
    const float* fLg    = (const float*)d_in[27];
    const float* fLb    = (const float*)d_in[28];
    (void)in_sizes; (void)n_in; (void)out_size;

    char* ws = (char*)d_ws;
    unsigned short* WT  = (unsigned short*)(ws + OFF_WT);
    float*          Af  = (float*)(ws + OFF_A);
    unsigned short* BQ  = (unsigned short*)(ws + OFF_BQ);
    unsigned short* BK  = (unsigned short*)(ws + OFF_BK);
    unsigned short* BV  = (unsigned short*)(ws + OFF_BV);
    float*          GT  = (float*)(ws + OFF_GT);
    float*          PB  = (float*)(ws + OFF_PB);
    float*          LG  = (float*)(ws + OFF_LG);
    unsigned short* AB  = (unsigned short*)(ws + OFF_AB);
    float*          OV  = (float*)(ws + OFF_OV);
    unsigned short* OG  = (unsigned short*)(ws + OFF_OG);
    float*          X1  = (float*)(ws + OFF_X1);
    float*          X2  = (float*)(ws + OFF_X2);
    unsigned short* XN  = (unsigned short*)(ws + OFF_XN);
    unsigned short* H   = (unsigned short*)(ws + OFF_A);   // alias: FFN hidden

    // transposed bf16 weight offsets (elements)
    unsigned short* WrK = WT + 0;
    unsigned short* WrQ = WT + 65536;
    unsigned short* WrV = WT + 131072;
    unsigned short* WrG = WT + 196608;
    unsigned short* WrO = WT + 262144;
    unsigned short* WcK = WT + 327680;
    unsigned short* WcQ = WT + 393216;
    unsigned short* WcV = WT + 458752;
    unsigned short* WcG = WT + 524288;
    unsigned short* WcO = WT + 589824;
    unsigned short* W1t = WT + 655360;   // 1024 x 256
    unsigned short* W2t = WT + 917504;   // 256 x 1024

    auto gemm = [&](const unsigned short* A, const unsigned short* B, void* Cp,
                    const float* bias, const float* add,
                    int M, int N, int K, int lda, int ldb, int ldc,
                    int act, int outbf, float scale, int batches, int b2,
                    long sA1, long sA2, long sB1, long sB2,
                    long sC1, long sC2, long sAdd1, long sAdd2) {
        GemmP p;
        p.A = A; p.B = B; p.C = Cp; p.bias = bias; p.add = add;
        p.sA1 = sA1; p.sA2 = sA2; p.sB1 = sB1; p.sB2 = sB2;
        p.sC1 = sC1; p.sC2 = sC2; p.sAdd1 = sAdd1; p.sAdd2 = sAdd2;
        p.M = M; p.N = N; p.K = K; p.lda = lda; p.ldb = ldb; p.ldc = ldc;
        p.act = act; p.b2 = b2; p.scale = scale;
        const int ntiles = (M / 32) * (N / 32);
        dim3 grid((ntiles + 7) / 8, batches);
        if (outbf) k_gemm<1><<<grid, 256, 0, stream>>>(p);
        else       k_gemm<0><<<grid, 256, 0, stream>>>(p);
    };

    const float invSqrtPPG = 0.17677669529663687f;   // 1/sqrt(32)
    const float invD       = 1.0f / 128.0f;

    // ---- weight prep (fp32 KxN -> bf16 NxK) ----
    k_wprep<<<256, 256, 0, stream>>>(rKw, WrK, 256, 256);
    k_wprep<<<256, 256, 0, stream>>>(rQw, WrQ, 256, 256);
    k_wprep<<<256, 256, 0, stream>>>(rVw, WrV, 256, 256);
    k_wprep<<<256, 256, 0, stream>>>(rGw, WrG, 256, 256);
    k_wprep<<<256, 256, 0, stream>>>(rOw, WrO, 256, 256);
    k_wprep<<<256, 256, 0, stream>>>(cKw, WcK, 256, 256);
    k_wprep<<<256, 256, 0, stream>>>(cQw, WcQ, 256, 256);
    k_wprep<<<256, 256, 0, stream>>>(cVw, WcV, 256, 256);
    k_wprep<<<256, 256, 0, stream>>>(cGw, WcG, 256, 256);
    k_wprep<<<256, 256, 0, stream>>>(cOw, WcO, 256, 256);
    k_wprep<<<1024, 256, 0, stream>>>(fW1, W1t, 256, 1024);
    k_wprep<<<1024, 256, 0, stream>>>(fW2, W2t, 1024, 256);

    // ================= row attention =================
    k_ln_bf16<<<TD, 256, 0, stream>>>(x, rLg, rLb, XN);
    k_ln_biasproj<<<LD * LD, 256, 0, stream>>>(biasT, rLbg, rLbb, rBw, PB);

    // projections (T x 256) x (256 x 256)
    gemm(XN, WrK, Af, nullptr, nullptr, TD, 256, 256, 256, 256, 256, 0, 0, 1.f,
         1, 1, 0,0,0,0,0,0,0,0);
    k_rearrange<<<4096, 256, 0, stream>>>(Af, BK, invSqrtPPG, 0);
    gemm(XN, WrQ, Af, nullptr, nullptr, TD, 256, 256, 256, 256, 256, 0, 0, 1.f,
         1, 1, 0,0,0,0,0,0,0,0);
    k_rearrange<<<4096, 256, 0, stream>>>(Af, BQ, invD, 0);
    gemm(XN, WrV, Af, nullptr, nullptr, TD, 256, 256, 256, 256, 256, 0, 0, 1.f,
         1, 1, 0,0,0,0,0,0,0,0);
    k_rearrange<<<4096, 256, 0, stream>>>(Af, BV, 1.f, 1);
    gemm(XN, WrG, GT, rGb, nullptr, TD, 256, 256, 256, 256, 256, 1, 0, 1.f,
         1, 1, 0,0,0,0,0,0,0,0);

    // logits[g][l][m] = Q[g] (384x4096) * K[g]^T + PB[g]
    gemm(BQ, BK, LG, nullptr, PB, 384, 384, 4096, 4096, 4096, 384, 0, 0, 1.f,
         8, 1, 384L*4096, 0, 384L*4096, 0, 384L*384, 0, 384L*384, 0);
    k_softmax_bf16<<<8 * 384, 256, 0, stream>>>(LG, AB, 384);
    // O[g][l][(d,c)] = attn[g] (384x384) * V^T[g] (4096x384)^T
    gemm(AB, BV, OV, nullptr, nullptr, 384, 4096, 384, 384, 384, 4096, 0, 0, 1.f,
         8, 1, 384L*384, 0, 4096L*384, 0, 384L*4096, 0, 0, 0);
    k_gate_mul<<<4096, 256, 0, stream>>>(GT, OV, OG);
    // out projection + residual
    gemm(OG, WrO, X1, rOb, x, TD, 256, 256, 256, 256, 256, 0, 0, 1.f,
         1, 1, 0,0,0,0,0,0,0,0);

    // ================= column attention =================
    k_ln_bf16<<<TD, 256, 0, stream>>>(X1, cLg, cLb, XN);
    gemm(XN, WcK, Af, nullptr, nullptr, TD, 256, 256, 256, 256, 256, 0, 0, 1.f,
         1, 1, 0,0,0,0,0,0,0,0);
    k_rearrange<<<4096, 256, 0, stream>>>(Af, BK, invSqrtPPG, 0);
    gemm(XN, WcQ, Af, nullptr, nullptr, TD, 256, 256, 256, 256, 256, 0, 0, 1.f,
         1, 1, 0,0,0,0,0,0,0,0);
    k_rearrange<<<4096, 256, 0, stream>>>(Af, BQ, 1.f, 0);
    gemm(XN, WcV, Af, nullptr, nullptr, TD, 256, 256, 256, 256, 256, 0, 0, 1.f,
         1, 1, 0,0,0,0,0,0,0,0);
    k_rearrange<<<4096, 256, 0, stream>>>(Af, BV, 1.f, 2);
    gemm(XN, WcG, GT, cGb, nullptr, TD, 256, 256, 256, 256, 256, 1, 0, 1.f,
         1, 1, 0,0,0,0,0,0,0,0);

    // chunk over l (8 chunks of 48) to bound logits buffer
    for (int lc = 0; lc < 8; ++lc) {
        const long l0 = (long)lc * 48;
        // logits[g][ll][d][k] = Q (128x32) * K^T per (g, l0+ll)
        gemm(BQ + l0 * 4096, BK + l0 * 4096, LG, nullptr, nullptr,
             128, 128, 32, 32, 32, 128, 0, 0, 1.f,
             8 * 48, 48, 384L*4096, 4096, 384L*4096, 4096,
             48L*16384, 16384, 0, 0);
        k_softmax_bf16<<<8 * 48 * 128, 256, 0, stream>>>(LG, AB, 128);
        // O[g][l0+ll][d][c] = attn (128x128) * Vt (32x128)^T
        gemm(AB, BV + l0 * 4096, OV + l0 * 4096, nullptr, nullptr,
             128, 32, 128, 128, 128, 32, 0, 0, 1.f,
             8 * 48, 48, 48L*16384, 16384, 384L*4096, 4096,
             384L*4096, 4096, 0, 0);
    }
    k_gate_mul<<<4096, 256, 0, stream>>>(GT, OV, OG);
    gemm(OG, WcO, X2, cOb, X1, TD, 256, 256, 256, 256, 256, 0, 0, 1.f,
         1, 1, 0,0,0,0,0,0,0,0);

    // ================= feed-forward =================
    k_ln_bf16<<<TD, 256, 0, stream>>>(X2, fLg, fLb, XN);
    gemm(XN, W1t, H, fB1, nullptr, TD, 1024, 256, 256, 256, 1024, 2, 1, 1.f,
         1, 1, 0,0,0,0,0,0,0,0);
    gemm(H, W2t, d_out, fB2, X2, TD, 256, 1024, 1024, 1024, 256, 0, 0, 1.f,
         1, 1, 0,0,0,0,0,0,0,0);
}
